// RNN_Cell_52192442581496
// MI455X (gfx1250) — compile-verified, matrix operand order checked by reading, not run on hardware
//
#include <hip/hip_runtime.h>
#include <hip/hip_bf16.h>

// ---------------------------------------------------------------------------
// RNN "gene" cell, persistent-kernel implementation for gfx1250 (MI455X).
// bf16 WMMA (v_wmma_f32_16x16x32_bf16), weights packed once into WMMA
// B-fragment order and kept L2-resident; device-wide barrier between
// dependency groups of the 9 per-step matmuls.
//
// Round-3 changes: x_t is staged to a bf16 double-buffer one step ahead
// (conversion folded into the writeout phase), so all matmul A/B operands
// are bf16 b128 loads — no f32->bf16 cvts in any K-loop; epilogue uses
// v_rcp/v_exp based sigmoid/tanh instead of ocml tanhf and precise division.
// ---------------------------------------------------------------------------

typedef __attribute__((ext_vector_type(16))) __bf16 v16bf;
typedef __attribute__((ext_vector_type(8)))  __bf16 v8bf;
typedef __attribute__((ext_vector_type(8)))  float  v8f;

#define BATCH   128
#define TSTEPS  256
#define NHID    1024
#define NINP    1024
#define NOUTW   2048            // every stage produces 2*NHID pre-gate columns
#define NWG     32              // persistent workgroups (256 waves, 8192 thr)

// ---- workspace layout (bytes) ----
#define OFF_BAR   0
#define OFF_PW0   1024
#define SZ_PW0    (8ull*1024*1024)                  // 2048x2048 bf16
#define OFF_PWS   (OFF_PW0 + SZ_PW0)
#define SZ_PWS    (32ull*1024*1024)                 // 8 x 1024x2048 bf16
#define OFF_H32   (OFF_PWS + SZ_PWS)
#define SZ_H32    (512ull*1024)
#define OFF_H16   (OFF_H32 + SZ_H32)
#define SZ_H16    (256ull*1024)
#define OFF_S32   (OFF_H16 + SZ_H16)
#define SZ_S32    (9ull*512*1024)
#define OFF_S16   (OFF_S32 + SZ_S32)
#define SZ_S16    (9ull*256*1024)
#define OFF_X16   (OFF_S16 + SZ_S16)                // 2 x 128x1024 bf16 staging
#define SZ_X16ONE (256ull*1024)

__device__ __forceinline__ v8f v8f_zero() {
    v8f z = {0.f,0.f,0.f,0.f,0.f,0.f,0.f,0.f};
    return z;
}

__device__ __forceinline__ v8f wmma_bf16(v16bf a, v16bf b, v8f c) {
    return __builtin_amdgcn_wmma_f32_16x16x32_bf16(
        /*neg_a=*/false, a, /*neg_b=*/false, b,
        /*c_mod=*/(short)0, c, /*reuse_a=*/false, /*reuse_b=*/false);
}

// Fast device activations (v_exp_f32 + v_rcp_f32; bf16-grade accuracy).
__device__ __forceinline__ float fast_sigmoid(float x) {
    return __builtin_amdgcn_rcpf(1.f + __expf(-x));
}
__device__ __forceinline__ float fast_tanh(float x) {
    return 2.f * __builtin_amdgcn_rcpf(1.f + __expf(-2.f * x)) - 1.f;
}
__device__ __forceinline__ float actf(int id, float x) {
    switch (id) {
        case 0:  return fast_sigmoid(x);
        case 1:  return fmaxf(x, 0.f);
        case 2:  return x;
        default: return fast_tanh(x);
    }
}

// A fragment (16x32 bf16) from a row-major bf16 matrix with row stride
// `stride` elements.  Per ISA layout: lane L -> row m = L%16, half h = L/16;
// elements 0..7 are cols k0+8h .. k0+8h+7, elements 8..15 cols k0+16+8h ..
__device__ __forceinline__ v16bf load_a_bf16(const __bf16* src, int stride,
                                             int m_base, int k0, int lane) {
    int m = m_base + (lane & 15);
    int h = lane >> 4;
    const __bf16* p = src + (size_t)m * stride + k0 + 8 * h;
    v8bf lo = *(const v8bf*)p;          // 16B aligned
    v8bf hi = *(const v8bf*)(p + 16);
    v16bf a;
#pragma unroll
    for (int i = 0; i < 8; ++i) { a[i] = lo[i]; a[8 + i] = hi[i]; }
    return a;
}

// B fragment from the pre-packed weight buffer: tile (n_tile, kt) stored as
// 32 lanes x 16 bf16 contiguously (1 KiB per tile); per-lane 32B load.
__device__ __forceinline__ v16bf load_b(const __bf16* packBase, int ktiles,
                                        int n_tile, int kt, int lane) {
    const __bf16* p = packBase + ((((size_t)n_tile * ktiles + kt) << 9) | ((size_t)lane << 4));
    return *(const v16bf*)p;
}

// ---- device-wide sense barrier over the NWG persistent workgroups ----
__device__ __forceinline__ void grid_sync(unsigned* bar) {
    __syncthreads();
    if (threadIdx.x == 0) {
        __threadfence();
        unsigned gen = __hip_atomic_load(&bar[1], __ATOMIC_RELAXED, __HIP_MEMORY_SCOPE_AGENT);
        unsigned prev = __hip_atomic_fetch_add(&bar[0], 1u, __ATOMIC_ACQ_REL, __HIP_MEMORY_SCOPE_AGENT);
        if (prev == (unsigned)(NWG - 1)) {
            __hip_atomic_store(&bar[0], 0u, __ATOMIC_RELAXED, __HIP_MEMORY_SCOPE_AGENT);
            __hip_atomic_fetch_add(&bar[1], 1u, __ATOMIC_RELEASE, __HIP_MEMORY_SCOPE_AGENT);
        } else {
            while (__hip_atomic_load(&bar[1], __ATOMIC_ACQUIRE, __HIP_MEMORY_SCOPE_AGENT) == gen) {
                __builtin_amdgcn_s_sleep(1);
            }
        }
        __threadfence();
    }
    __syncthreads();
}

// ---------------------------------------------------------------------------
// Fragment bundle for one pipeline slot: two A row-tiles + two B col-tiles.
// ---------------------------------------------------------------------------
struct Frag { v16bf a0, a1, bc, bh; };

__device__ __forceinline__ void load_frags(Frag& f,
    const __bf16* aSrc, int aStride,
    const __bf16* pw, int ktilesB, int m_base, int n_tile,
    int kA, int kB, int lane)
{
    f.a0 = load_a_bf16(aSrc, aStride, m_base,      kA * 32, lane);
    f.a1 = load_a_bf16(aSrc, aStride, m_base + 16, kA * 32, lane);
    f.bc = load_b(pw, ktilesB, n_tile,      kB, lane);
    f.bh = load_b(pw, ktilesB, n_tile + 64, kB, lane);
}

// KT k-steps of the 4-chain matmul, double-buffered (unroll 2).  The final
// preload is clamped in-bounds and its data is never consumed.
template<int KT>
__device__ __forceinline__ void mm_block(
    const __bf16* aSrc, int aStride,
    const __bf16* pw, int ktilesB, int kBoff,
    int m_base, int n_tile, int lane,
    v8f& accC0, v8f& accC1, v8f& accH0, v8f& accH1)
{
    Frag f0, f1;
    load_frags(f0, aSrc, aStride, pw, ktilesB, m_base, n_tile, 0, kBoff, lane);
#pragma unroll 4
    for (int kt = 0; kt < KT; kt += 2) {
        load_frags(f1, aSrc, aStride, pw, ktilesB, m_base, n_tile,
                   kt + 1, kBoff + kt + 1, lane);
        accC0 = wmma_bf16(f0.a0, f0.bc, accC0);
        accH0 = wmma_bf16(f0.a0, f0.bh, accH0);
        accC1 = wmma_bf16(f0.a1, f0.bc, accC1);
        accH1 = wmma_bf16(f0.a1, f0.bh, accH1);
        int kn = (kt + 2 < KT) ? kt + 2 : kt;      // dead tail preload, in-bounds
        load_frags(f0, aSrc, aStride, pw, ktilesB, m_base, n_tile,
                   kn, kBoff + kn, lane);
        accC0 = wmma_bf16(f1.a0, f1.bc, accC0);
        accH0 = wmma_bf16(f1.a0, f1.bh, accH0);
        accC1 = wmma_bf16(f1.a1, f1.bc, accC1);
        accH1 = wmma_bf16(f1.a1, f1.bh, accH1);
    }
}

// Gating epilogue for one 16x16 C/D tile pair (c-gate + h-gate columns).
// C/D layout: VGPR v -> m = v + 8*(lane>>4), n = lane&15.
__device__ __forceinline__ void epi_tile(const v8f& accC, const v8f& accH, int actId,
                                         const float* __restrict__ prevBuf,
                                         float* __restrict__ d32,
                                         __bf16* __restrict__ d16,
                                         int mrow, int n, v8f& sum) {
#pragma unroll
    for (int v = 0; v < 8; ++v) {
        size_t idx = (size_t)(mrow + v) * NHID + n;
        float prev = prevBuf[idx];
        float c  = fast_sigmoid(accC[v]);
        float hh = actf(actId, accH[v]);
        float sn = prev + c * (hh - prev);
        d32[idx] = sn;
        d16[idx] = (__bf16)sn;
        sum[v] += sn;
    }
}

// ---------------------------------------------------------------------------
// Prep kernel 1: convert W0 and Ws (fp32 row-major K x 2048) into bf16
// WMMA B-fragment order.  One thread per packed bf16 element.
// ---------------------------------------------------------------------------
__global__ void pack_weights_kernel(const float* __restrict__ W0,
                                    const float* __restrict__ Ws,
                                    __bf16* __restrict__ pW0,
                                    __bf16* __restrict__ pWs) {
    size_t gid = (size_t)blockIdx.x * blockDim.x + threadIdx.x;
    const size_t TOT0 = 128ull * 64 * 512;     // W0: 128 n-tiles, 64 k-tiles
    const size_t TOTS = 8ull * 128 * 32 * 512; // Ws: 8 layers, 32 k-tiles
    if (gid < TOT0) {
        size_t tile = gid >> 9; int r = (int)(gid & 511);
        int lane = r >> 4, e = r & 15;
        int n_tile = (int)(tile >> 6), k_tile = (int)(tile & 63);
        int n = n_tile * 16 + (lane & 15);
        int h = lane >> 4;
        int p = e >> 1, w = e & 1;
        int kl = (p < 4 ? 2 * p : 16 + 2 * (p - 4)) + 8 * h + w;
        int k = k_tile * 32 + kl;
        pW0[gid] = (__bf16)W0[(size_t)k * NOUTW + n];
    } else if (gid < TOT0 + TOTS) {
        size_t g = gid - TOT0;
        const size_t per_layer = 128ull * 32 * 512;
        int l = (int)(g / per_layer);
        size_t r2 = g % per_layer;
        size_t tile = r2 >> 9; int r = (int)(r2 & 511);
        int lane = r >> 4, e = r & 15;
        int n_tile = (int)(tile >> 5), k_tile = (int)(tile & 31);
        int n = n_tile * 16 + (lane & 15);
        int h = lane >> 4;
        int p = e >> 1, w = e & 1;
        int kl = (p < 4 ? 2 * p : 16 + 2 * (p - 4)) + 8 * h + w;
        int k = k_tile * 32 + kl;
        pWs[g] = (__bf16)Ws[((size_t)l * NHID + k) * NOUTW + n];
    }
}

// Prep kernel 2: seed hidden state (f32 + bf16 mirror), stage x_0 to bf16,
// and reset the barrier.
__global__ void init_state_kernel(const float* __restrict__ hidden,
                                  const float* __restrict__ x,
                                  float* __restrict__ h32,
                                  __bf16* __restrict__ h16,
                                  __bf16* __restrict__ x16a,
                                  unsigned* __restrict__ bar) {
    int gid = blockIdx.x * blockDim.x + threadIdx.x;
    if (gid < BATCH * NHID) {
        float v = hidden[gid];
        h32[gid] = v;
        h16[gid] = (__bf16)v;
        int m = gid >> 10, c = gid & 1023;
        x16a[gid] = (__bf16)x[((size_t)m * TSTEPS + 0) * NINP + c];
    }
    if (gid == 0) { bar[0] = 0u; bar[1] = 0u; }
}

// ---------------------------------------------------------------------------
// One gene stage for this wave's (2 m-tiles) x (n_tile, n_tile+64) block.
// ---------------------------------------------------------------------------
__device__ __forceinline__ void run_gene(int g, int inI, int actId,
                                         const __bf16* __restrict__ pWs,
                                         float* __restrict__ s32,
                                         __bf16* __restrict__ s16,
                                         int m_base, int n_tile, int lane,
                                         v8f& sum0, v8f& sum1) {
    const __bf16* pw   = pWs + (size_t)g * (128ull * 32 * 512);
    const __bf16* aSrc = s16 + (size_t)inI * (BATCH * NHID);
    const float*  pSrc = s32 + (size_t)inI * (BATCH * NHID);
    float*        d32  = s32 + (size_t)(g + 1) * (BATCH * NHID);
    __bf16*       d16  = s16 + (size_t)(g + 1) * (BATCH * NHID);

    v8f accC0 = v8f_zero(), accC1 = v8f_zero();
    v8f accH0 = v8f_zero(), accH1 = v8f_zero();
    mm_block<32>(aSrc, NHID, pw, 32, 0, m_base, n_tile, lane,
                 accC0, accC1, accH0, accH1);

    int n     = n_tile * 16 + (lane & 15);
    int mrow0 = m_base + 8 * (lane >> 4);
    epi_tile(accC0, accH0, actId, pSrc, d32, d16, mrow0,      n, sum0);
    epi_tile(accC1, accH1, actId, pSrc, d32, d16, mrow0 + 16, n, sum1);
}

// ---------------------------------------------------------------------------
// Persistent main kernel: 32 WGs x 256 threads = 256 waves.
// Wave mapping (for WGP$ reuse):  wid = tid>>5;
//   m_base = (wid & 3) * 32  -> waves sharing n stream differ in m (B reuse)
//   n_tile = blockIdx.x * 2 + (wid >> 2)
// Each wave owns rows [m_base, m_base+32) x columns [n, n+16) of the state
// for every stage of every timestep; the running mean lives in 16 VGPRs.
// Schedule: init | g0 | g1,g2,g3 | g4,g6 | g5,g7  (5 grid syncs / step).
// x_{t+1} is converted fp32->bf16 into a double buffer during step t.
// ---------------------------------------------------------------------------
__global__ __launch_bounds__(256) void rnn_persistent_kernel(
    const float* __restrict__ x,      // (B, T, NINP)
    float* __restrict__ out,          // (B, T, NHID) ++ (B, NHID)
    const __bf16* __restrict__ pW0,
    const __bf16* __restrict__ pWs,
    float* __restrict__ h32, __bf16* __restrict__ h16,
    float* __restrict__ s32, __bf16* __restrict__ s16,
    __bf16* __restrict__ x16,         // 2 staging buffers of BATCH*NINP
    unsigned* __restrict__ bar) {

    const int lane   = threadIdx.x & 31;
    const int wid    = threadIdx.x >> 5;             // 0..7
    const int m_base = (wid & 3) * 32;               // 0,32,64,96
    const int n_tile = blockIdx.x * 2 + (wid >> 2);  // 0..63
    const int gtid   = blockIdx.x * 256 + threadIdx.x; // 0..8191

    v8f sum0 = v8f_zero(), sum1 = v8f_zero();

    for (int t = 0; t < TSTEPS; ++t) {
        const __bf16* xcur = x16 + (size_t)(t & 1) * (BATCH * NINP);
        __bf16*       xnxt = x16 + (size_t)((t + 1) & 1) * (BATCH * NINP);

        // ---- stage init: ch = [x_t, h_prev] @ W0 (K = 2048, split loops) ----
        {
            v8f accC0 = v8f_zero(), accC1 = v8f_zero();
            v8f accH0 = v8f_zero(), accH1 = v8f_zero();
            mm_block<32>(xcur, NINP, pW0, 64, 0,
                         m_base, n_tile, lane, accC0, accC1, accH0, accH1);
            mm_block<32>(h16, NHID, pW0, 64, 32,
                         m_base, n_tile, lane, accC0, accC1, accH0, accH1);

            int n     = n_tile * 16 + (lane & 15);
            int mrow0 = m_base + 8 * (lane >> 4);
            sum0 = v8f_zero(); sum1 = v8f_zero();
            epi_tile(accC0, accH0, /*tanh*/3, h32, s32, s16, mrow0,      n, sum0);
            epi_tile(accC1, accH1, /*tanh*/3, h32, s32, s16, mrow0 + 16, n, sum1);
            sum0 = v8f_zero(); sum1 = v8f_zero();   // s0 not part of the mean
        }
        grid_sync(bar);

        run_gene(0, 0, 0, pWs, s32, s16, m_base, n_tile, lane, sum0, sum1); // s1<-s0 sig
        grid_sync(bar);
        run_gene(1, 1, 1, pWs, s32, s16, m_base, n_tile, lane, sum0, sum1); // s2<-s1 relu
        run_gene(2, 1, 1, pWs, s32, s16, m_base, n_tile, lane, sum0, sum1); // s3<-s1 relu
        run_gene(3, 1, 2, pWs, s32, s16, m_base, n_tile, lane, sum0, sum1); // s4<-s1 id
        grid_sync(bar);
        run_gene(4, 2, 3, pWs, s32, s16, m_base, n_tile, lane, sum0, sum1); // s5<-s2 tanh
        run_gene(6, 3, 3, pWs, s32, s16, m_base, n_tile, lane, sum0, sum1); // s7<-s3 tanh
        grid_sync(bar);
        run_gene(5, 5, 0, pWs, s32, s16, m_base, n_tile, lane, sum0, sum1); // s6<-s5 sig
        run_gene(7, 5, 1, pWs, s32, s16, m_base, n_tile, lane, sum0, sum1); // s8<-s5 relu

        // ---- write h_new = mean(s1..s8); update state; stage x_{t+1} ----
        {
            int n     = n_tile * 16 + (lane & 15);
            int mrow0 = m_base + 8 * (lane >> 4);
#pragma unroll
            for (int v = 0; v < 8; ++v) {
#pragma unroll
                for (int half = 0; half < 2; ++half) {
                    int m = mrow0 + 16 * half + v;
                    float hn = (half ? sum1[v] : sum0[v]) * 0.125f;
                    out[((size_t)m * TSTEPS + t) * NHID + n] = hn;
                    size_t idx = (size_t)m * NHID + n;
                    h32[idx] = hn;
                    h16[idx] = (__bf16)hn;
                    if (t == TSTEPS - 1)
                        out[(size_t)BATCH * TSTEPS * NHID + idx] = hn;
                }
            }
            // convert x_{t+1}: 131072 elements over 8192 threads, 16 each
            if (t + 1 < TSTEPS) {
                int base = gtid * 16;
                int m = base >> 10, c = base & 1023;
                const float* src = x + ((size_t)m * TSTEPS + (t + 1)) * NINP + c;
#pragma unroll
                for (int i = 0; i < 16; ++i)
                    xnxt[(size_t)m * NINP + c + i] = (__bf16)src[i];
            }
        }
        grid_sync(bar);
    }
}

// ---------------------------------------------------------------------------
extern "C" void kernel_launch(void* const* d_in, const int* in_sizes, int n_in,
                              void* d_out, int out_size, void* d_ws, size_t ws_size,
                              hipStream_t stream) {
    (void)in_sizes; (void)n_in; (void)out_size; (void)ws_size;
    const float* x   = (const float*)d_in[0];   // (128,256,1024)
    const float* hid = (const float*)d_in[1];   // (1,128,1024)
    const float* W0  = (const float*)d_in[2];   // (2048,2048)
    const float* Ws  = (const float*)d_in[3];   // (8,1024,2048)
    float* out = (float*)d_out;

    char* ws = (char*)d_ws;
    unsigned* bar = (unsigned*)(ws + OFF_BAR);
    __bf16*   pW0 = (__bf16*)(ws + OFF_PW0);
    __bf16*   pWs = (__bf16*)(ws + OFF_PWS);
    float*    h32 = (float*) (ws + OFF_H32);
    __bf16*   h16 = (__bf16*)(ws + OFF_H16);
    float*    s32 = (float*) (ws + OFF_S32);
    __bf16*   s16 = (__bf16*)(ws + OFF_S16);
    __bf16*   x16 = (__bf16*)(ws + OFF_X16);

    // 20,971,520 packed weight elements / 256 threads = 81920 blocks
    pack_weights_kernel<<<81920, 256, 0, stream>>>(W0, Ws, pW0, pWs);
    init_state_kernel<<<512, 256, 0, stream>>>(hid, x, h32, h16, x16, bar);
    rnn_persistent_kernel<<<NWG, 256, 0, stream>>>(x, out, pW0, pWs,
                                                   h32, h16, s32, s16, x16, bar);
}
